// TemporalAttention_81999515615745
// MI455X (gfx1250) — matrix operand order repaired and checked by reading, not compiled, not run on hardware
//
#include <hip/hip_runtime.h>

// Problem sizes (fixed by reference)
#define B_  4
#define S_  2048
#define F_  512
#define H_  512
#define NH_ 8
#define HD_ 64
#define SSTR 2052   // padded LDS score row stride (floats); 8*2052 % 64 != 0 -> no cross-half bank clash

typedef __bf16 bf16;
typedef __attribute__((ext_vector_type(16))) __bf16 v16bf;
typedef __attribute__((ext_vector_type(8)))  __bf16 v8bf;
typedef __attribute__((ext_vector_type(8)))  float  v8f;
typedef __attribute__((ext_vector_type(4)))  float  vf4;

#define DEV static __device__ __forceinline__

DEV v8f wmma_bf16(v16bf a, v16bf b, v8f c) {
    // D(16x16,f32) = A(16x32,bf16) * B(32x16,bf16) + C
    return __builtin_amdgcn_wmma_f32_16x16x32_bf16(false, a, false, b, (short)0, c, false, false);
}

// A-frag: row-major MxK bf16, lane m=lane&15 holds K = {k0+half*8 .. +7} and {k0+16+half*8 .. +7}
DEV v16bf load_frag_a_bf16(const bf16* __restrict__ base, int lda, int k0) {
    const int lane = threadIdx.x & 31;
    const int m = lane & 15, hf = lane >> 4;
    const bf16* p = base + m * lda + k0 + hf * 8;
    v8bf lo = *(const v8bf*)(p);
    v8bf hi = *(const v8bf*)(p + 16);
    v16bf r;
#pragma unroll
    for (int i = 0; i < 8; ++i) { r[i] = lo[i]; r[i + 8] = hi[i]; }
    return r;
}

// B-frag: B[k][n] = W[n][k] with W row-major NxK; lane n=lane&15 holds K = {k0+half*8 .. +7} and {k0+16+half*8 .. +7}
DEV v16bf load_frag_b_bf16(const bf16* __restrict__ base, int ldb, int k0) {
    const int lane = threadIdx.x & 31;
    const int n = lane & 15, hf = lane >> 4;
    const bf16* p = base + n * ldb + k0 + hf * 8;
    v8bf lo = *(const v8bf*)(p);
    v8bf hi = *(const v8bf*)(p + 16);
    v16bf r;
#pragma unroll
    for (int i = 0; i < 8; ++i) { r[i] = lo[i]; r[i + 8] = hi[i]; }
    return r;
}

// A-frag built from f32 LDS scores with on-the-fly bf16 convert
DEV v16bf load_frag_a_f32(const float* __restrict__ base, int lda, int k0) {
    const int lane = threadIdx.x & 31;
    const int m = lane & 15, hf = lane >> 4;
    const float* p = base + m * lda + k0 + hf * 8;
    vf4 x0 = *(const vf4*)(p);
    vf4 x1 = *(const vf4*)(p + 4);
    vf4 x2 = *(const vf4*)(p + 16);
    vf4 x3 = *(const vf4*)(p + 20);
    v16bf r;
    r[0]=(bf16)x0.x; r[1]=(bf16)x0.y; r[2]=(bf16)x0.z; r[3]=(bf16)x0.w;
    r[4]=(bf16)x1.x; r[5]=(bf16)x1.y; r[6]=(bf16)x1.z; r[7]=(bf16)x1.w;
    r[8]=(bf16)x2.x; r[9]=(bf16)x2.y; r[10]=(bf16)x2.z; r[11]=(bf16)x2.w;
    r[12]=(bf16)x3.x; r[13]=(bf16)x3.y; r[14]=(bf16)x3.z; r[15]=(bf16)x3.w;
    return r;
}

// ---------------------------------------------------------------- cvt f32->bf16
__global__ void cvt_f32_bf16(const float* __restrict__ src, bf16* __restrict__ dst, int n) {
    int i = (blockIdx.x * blockDim.x + threadIdx.x) * 4;
    if (i + 3 < n) {
        vf4 v = *(const vf4*)(src + i);
        dst[i + 0] = (bf16)v.x; dst[i + 1] = (bf16)v.y;
        dst[i + 2] = (bf16)v.z; dst[i + 3] = (bf16)v.w;
    } else {
        for (int k = 0; k < 4 && i + k < n; ++k) dst[i + k] = (bf16)src[i + k];
    }
}

// ---------------------------------------------------------------- QKV projection
// M=B*S=8192, N=H=512, K=F=512. Epilogue scatters into the bug-compatible
// head-scrambled layout: head=s>>8, s'=((s&255)<<3)|(h>>6), d=h&63.
// Q,K -> [b,head,s',d] row-major; V -> transposed [b,head,d,s'].
__global__ void qkv_proj_kernel(
    const bf16* __restrict__ seqb,
    const bf16* __restrict__ Wqb, const bf16* __restrict__ Wkb, const bf16* __restrict__ Wvb,
    const float* __restrict__ bq, const float* __restrict__ bk, const float* __restrict__ bv,
    bf16* __restrict__ Qs, bf16* __restrict__ Ks, bf16* __restrict__ Vt)
{
    const int z = blockIdx.z;
    const bf16*  W    = (z == 0) ? Wqb : (z == 1) ? Wkb : Wvb;
    const float* bias = (z == 0) ? bq  : (z == 1) ? bk  : bv;

    const int wave = threadIdx.x >> 5;
    const int m0 = blockIdx.x * 32 + (wave >> 1) * 16;   // row in [0, 8192)
    const int n0 = blockIdx.y * 32 + (wave & 1) * 16;    // col in [0, 512)

    v8f acc = {};
    for (int k = 0; k < F_; k += 32) {
        v16bf a = load_frag_a_bf16(seqb + (size_t)m0 * F_, F_, k);
        v16bf w = load_frag_b_bf16(W    + (size_t)n0 * F_, F_, k);
        acc = wmma_bf16(a, w, acc);
    }

    const int lane = threadIdx.x & 31;
    const int n = lane & 15, hf = lane >> 4;
    const int col = n0 + n;
    const float bv_ = bias[col];
    const int d  = col & 63;
    const int hh = col >> 6;
#pragma unroll
    for (int j = 0; j < 8; ++j) {
        const int mg = m0 + j + 8 * hf;
        const int bidx = mg >> 11;          // batch
        const int s    = mg & 2047;         // sequence position
        const int head = s >> 8;
        const int sp   = ((s & 255) << 3) | hh;
        const bf16 v = (bf16)(acc[j] + bv_);
        if (z == 2) {
            Vt[((size_t)(bidx * NH_ + head) * HD_ + d) * S_ + sp] = v;
        } else {
            bf16* dst = (z == 0) ? Qs : Ks;
            dst[((size_t)(bidx * NH_ + head) * S_ + sp) * HD_ + d] = v;
        }
    }
}

// ---------------------------------------------------------------- fused attention
// One block per (b, head, 16-row q-block). 256 threads = 8 waves.
// Phase 1: scores[16][2048] = (Q Kt)/8 via WMMA -> LDS (f32, padded stride).
// Phase 2: mask-as-zero + softmax in LDS; attn f32 streamed to d_out (NT stores).
// Phase 3: attn(bf16 cvt from LDS) @ V via WMMA, K-split over 8 waves + LDS reduce.
__global__ void attn_fused_kernel(
    const bf16* __restrict__ Qs, const bf16* __restrict__ Ks, const bf16* __restrict__ Vt,
    const int*  __restrict__ mask,
    float* __restrict__ attn_out,        // [B,NH,S,S]
    bf16*  __restrict__ att2)            // [B,S,H] (head-major cols)
{
    extern __shared__ float smem[];
    float* scores  = smem;                       // [16][SSTR]
    float* partial = smem + 16 * SSTR;           // [4][16][16] = 1024 f32
    float* redbuf  = partial + 1024;             // [16][16]    = 256  f32

    const int qb = blockIdx.x;                   // 0..127
    const int h  = blockIdx.y;                   // 0..7
    const int b  = blockIdx.z;                   // 0..3
    const int qg = qb * 16;

    const int tid  = threadIdx.x;
    const int wave = tid >> 5;
    const int lane = tid & 31;
    const int ln = lane & 15, hf = lane >> 4;

    const bf16* Qbh = Qs + ((size_t)(b * NH_ + h) * S_ + qg) * HD_;
    const bf16* Kbh = Ks + ((size_t)(b * NH_ + h) * S_) * HD_;
    const bf16* Vbh = Vt + ((size_t)(b * NH_ + h) * HD_) * S_;

    // ---- Phase 1: scores strip
    {
        v16bf a0 = load_frag_a_bf16(Qbh, HD_, 0);
        v16bf a1 = load_frag_a_bf16(Qbh, HD_, 32);
        for (int t = 0; t < 16; ++t) {
            const int nt = wave + 8 * t;
            v16bf b0 = load_frag_b_bf16(Kbh + (size_t)nt * 16 * HD_, HD_, 0);
            v16bf b1 = load_frag_b_bf16(Kbh + (size_t)nt * 16 * HD_, HD_, 32);
            v8f acc = {};
            acc = wmma_bf16(a0, b0, acc);
            acc = wmma_bf16(a1, b1, acc);
#pragma unroll
            for (int j = 0; j < 8; ++j)
                scores[(j + 8 * hf) * SSTR + nt * 16 + ln] = acc[j] * 0.125f;
        }
    }
    __syncthreads();

    // ---- Phase 2: softmax (16 threads per row, 128 cols each as 16x8-chunks)
    {
        const int r  = tid >> 4;
        const int jj = tid & 15;
        float* row = scores + r * SSTR;
        const int* mrow = mask + b * S_;
        float* arow = attn_out + ((size_t)(b * NH_ + h) * S_ + (qg + r)) * (size_t)S_;

        float mx = -3.0e38f;
        for (int i = 0; i < 16; ++i) {
            const int c = jj * 8 + i * 128;
            vf4 v0 = *(vf4*)(row + c);
            vf4 v1 = *(vf4*)(row + c + 4);
            const int4 q0 = *(const int4*)(mrow + c);
            const int4 q1 = *(const int4*)(mrow + c + 4);
            v0.x = q0.x ? v0.x : 0.0f; v0.y = q0.y ? v0.y : 0.0f;
            v0.z = q0.z ? v0.z : 0.0f; v0.w = q0.w ? v0.w : 0.0f;
            v1.x = q1.x ? v1.x : 0.0f; v1.y = q1.y ? v1.y : 0.0f;
            v1.z = q1.z ? v1.z : 0.0f; v1.w = q1.w ? v1.w : 0.0f;
            *(vf4*)(row + c)     = v0;
            *(vf4*)(row + c + 4) = v1;
            mx = fmaxf(mx, fmaxf(fmaxf(v0.x, v0.y), fmaxf(v0.z, v0.w)));
            mx = fmaxf(mx, fmaxf(fmaxf(v1.x, v1.y), fmaxf(v1.z, v1.w)));
        }
        redbuf[r * 16 + jj] = mx;
        __syncthreads();
        float rmax = redbuf[r * 16];
#pragma unroll
        for (int i = 1; i < 16; ++i) rmax = fmaxf(rmax, redbuf[r * 16 + i]);
        __syncthreads();

        float sum = 0.0f;
        for (int i = 0; i < 16; ++i) {
            const int c = jj * 8 + i * 128;
            vf4 v0 = *(vf4*)(row + c);
            vf4 v1 = *(vf4*)(row + c + 4);
            v0.x = __expf(v0.x - rmax); v0.y = __expf(v0.y - rmax);
            v0.z = __expf(v0.z - rmax); v0.w = __expf(v0.w - rmax);
            v1.x = __expf(v1.x - rmax); v1.y = __expf(v1.y - rmax);
            v1.z = __expf(v1.z - rmax); v1.w = __expf(v1.w - rmax);
            *(vf4*)(row + c)     = v0;
            *(vf4*)(row + c + 4) = v1;
            sum += (v0.x + v0.y) + (v0.z + v0.w) + (v1.x + v1.y) + (v1.z + v1.w);
        }
        redbuf[r * 16 + jj] = sum;
        __syncthreads();
        float rsum = 0.0f;
#pragma unroll
        for (int i = 0; i < 16; ++i) rsum += redbuf[r * 16 + i];
        const float inv = 1.0f / rsum;

        for (int i = 0; i < 16; ++i) {
            const int c = jj * 8 + i * 128;
            vf4 v0 = *(vf4*)(row + c);
            vf4 v1 = *(vf4*)(row + c + 4);
            v0.x *= inv; v0.y *= inv; v0.z *= inv; v0.w *= inv;
            v1.x *= inv; v1.y *= inv; v1.z *= inv; v1.w *= inv;
            *(vf4*)(row + c)     = v0;
            *(vf4*)(row + c + 4) = v1;
            __builtin_nontemporal_store(v0, (vf4*)(arow + c));      // streaming output
            __builtin_nontemporal_store(v1, (vf4*)(arow + c + 4));
        }
    }
    __syncthreads();

    // ---- Phase 3: attended[16 x 64] = attn[16 x 2048] @ V[2048 x 64]
    {
        const int nt3 = wave & 3;      // HD tile (16 cols)
        const int kh  = wave >> 2;     // K half
        const bf16* Vtile = Vbh + (size_t)nt3 * 16 * S_;
        v8f acc = {};
        const int kbeg = kh * 1024;
        for (int kk = kbeg; kk < kbeg + 1024; kk += 32) {
            v16bf a  = load_frag_a_f32(scores, SSTR, kk);
            v16bf vb = load_frag_b_bf16(Vtile, S_, kk);
            acc = wmma_bf16(a, vb, acc);
        }
        if (kh == 1) {
#pragma unroll
            for (int j = 0; j < 8; ++j)
                partial[(nt3 * 16 + j + 8 * hf) * 16 + ln] = acc[j];
        }
        __syncthreads();
        if (kh == 0) {
#pragma unroll
            for (int j = 0; j < 8; ++j)
                acc[j] += partial[(nt3 * 16 + j + 8 * hf) * 16 + ln];
#pragma unroll
            for (int j = 0; j < 8; ++j) {
                const int m = j + 8 * hf;   // q row within block
                att2[(size_t)(b * S_ + qg + m) * H_ + h * HD_ + nt3 * 16 + ln] = (bf16)acc[j];
            }
        }
    }
}

// ---------------------------------------------------------------- output projection
__global__ void out_proj_kernel(const bf16* __restrict__ att2, const bf16* __restrict__ Wob,
                                const float* __restrict__ bo, float* __restrict__ out)
{
    const int wave = threadIdx.x >> 5;
    const int m0 = blockIdx.x * 32 + (wave >> 1) * 16;
    const int n0 = blockIdx.y * 32 + (wave & 1) * 16;
    v8f acc = {};
    for (int k = 0; k < H_; k += 32) {
        v16bf a = load_frag_a_bf16(att2 + (size_t)m0 * H_, H_, k);
        v16bf w = load_frag_b_bf16(Wob  + (size_t)n0 * H_, H_, k);
        acc = wmma_bf16(a, w, acc);
    }
    const int lane = threadIdx.x & 31;
    const int n = lane & 15, hf = lane >> 4;
    const float bv_ = bo[n0 + n];
#pragma unroll
    for (int j = 0; j < 8; ++j)
        out[(size_t)(m0 + j + 8 * hf) * H_ + n0 + n] = acc[j] + bv_;
}

// ---------------------------------------------------------------- host launcher
extern "C" void kernel_launch(void* const* d_in, const int* in_sizes, int n_in,
                              void* d_out, int out_size, void* d_ws, size_t ws_size,
                              hipStream_t stream) {
    (void)in_sizes; (void)n_in; (void)out_size; (void)ws_size;

    const float* seq  = (const float*)d_in[0];
    const int*   mask = (const int*)  d_in[1];
    const float* Wq   = (const float*)d_in[2];
    const float* bq   = (const float*)d_in[3];
    const float* Wk   = (const float*)d_in[4];
    const float* bk   = (const float*)d_in[5];
    const float* Wv   = (const float*)d_in[6];
    const float* bv   = (const float*)d_in[7];
    const float* Wo   = (const float*)d_in[8];
    const float* bo   = (const float*)d_in[9];

    float* out  = (float*)d_out;                       // [B,S,H]
    float* attn = out + (size_t)B_ * S_ * H_;          // [B,NH,S,S]

    char* ws = (char*)d_ws;
    const size_t SEQ_B  = (size_t)B_ * S_ * F_ * 2;    // 8 MB
    const size_t W_B    = (size_t)H_ * F_ * 2;         // 512 KB
    const size_t QKV_B  = (size_t)B_ * NH_ * S_ * HD_ * 2;  // 8 MB

    bf16* seqb = (bf16*)(ws);
    bf16* Wqb  = (bf16*)(ws + SEQ_B);
    bf16* Wkb  = (bf16*)(ws + SEQ_B + 1 * W_B);
    bf16* Wvb  = (bf16*)(ws + SEQ_B + 2 * W_B);
    bf16* Wob  = (bf16*)(ws + SEQ_B + 3 * W_B);
    bf16* Qs   = (bf16*)(ws + SEQ_B + 4 * W_B);
    bf16* Ks   = (bf16*)(ws + SEQ_B + 4 * W_B + 1 * QKV_B);
    bf16* Vt   = (bf16*)(ws + SEQ_B + 4 * W_B + 2 * QKV_B);
    bf16* att2 = (bf16*)(ws + SEQ_B + 4 * W_B + 3 * QKV_B);

    // 1) convert inputs to bf16
    {
        const int nseq = B_ * S_ * F_;
        cvt_f32_bf16<<<dim3(nseq / (4 * 256)), 256, 0, stream>>>(seq, seqb, nseq);
        const int nw = H_ * F_;
        cvt_f32_bf16<<<dim3(nw / (4 * 256)), 256, 0, stream>>>(Wq, Wqb, nw);
        cvt_f32_bf16<<<dim3(nw / (4 * 256)), 256, 0, stream>>>(Wk, Wkb, nw);
        cvt_f32_bf16<<<dim3(nw / (4 * 256)), 256, 0, stream>>>(Wv, Wvb, nw);
        cvt_f32_bf16<<<dim3(nw / (4 * 256)), 256, 0, stream>>>(Wo, Wob, nw);
    }

    // 2) QKV projections (M=8192 rows, N=512 cols, z selects Q/K/V)
    qkv_proj_kernel<<<dim3((B_ * S_) / 32, H_ / 32, 3), 128, 0, stream>>>(
        seqb, Wqb, Wkb, Wvb, bq, bk, bv, Qs, Ks, Vt);

    // 3) fused scores -> softmax -> attn@V  (dynamic LDS: 16*2052 + 1024 + 256 f32)
    const int smem_bytes = (16 * SSTR + 1024 + 256) * (int)sizeof(float);  // 136448
    (void)hipFuncSetAttribute((const void*)attn_fused_kernel,
                              hipFuncAttributeMaxDynamicSharedMemorySize, smem_bytes);
    attn_fused_kernel<<<dim3(S_ / 16, NH_, B_), 256, smem_bytes, stream>>>(
        Qs, Ks, Vt, mask, attn, att2);

    // 4) output projection
    out_proj_kernel<<<dim3((B_ * S_) / 32, H_ / 32, 1), 128, 0, stream>>>(
        att2, Wob, bo, out);
}
